// FasterRCNNTrainer_56848187129931
// MI455X (gfx1250) — matrix operand order repaired and checked by reading, not compiled, not run on hardware
//
#include <hip/hip_runtime.h>

// ---------------------------------------------------------------------------
// CDNA5 (gfx1250) types for WMMA
// ---------------------------------------------------------------------------
typedef __attribute__((ext_vector_type(16))) __bf16 v16bf;
typedef __attribute__((ext_vector_type(8)))  __bf16 v8bf;
typedef __attribute__((ext_vector_type(8)))  float  v8f;

__device__ __forceinline__ unsigned short f2bf(float f) {
    // round-to-nearest-even f32 -> bf16
    unsigned u = __builtin_bit_cast(unsigned, f);
    u += 0x7FFFu + ((u >> 16) & 1u);
    return (unsigned short)(u >> 16);
}

// pack two f32 into one dword of bf16 (lo = f0, hi = f1), RNE rounding
__device__ __forceinline__ unsigned pack2bf(float f0, float f1) {
    unsigned u0 = __builtin_bit_cast(unsigned, f0);
    unsigned u1 = __builtin_bit_cast(unsigned, f1);
    u0 += 0x7FFFu + ((u0 >> 16) & 1u);
    u1 += 0x7FFFu + ((u1 >> 16) & 1u);
    return (u0 >> 16) | (u1 & 0xFFFF0000u);
}

// ---------------------------------------------------------------------------
// Problem constants
// ---------------------------------------------------------------------------
#define HIMG 608
#define WIMG 1008
#define FH   38
#define FW   63
#define NPOS (FH * FW)          // 2394
#define NANCH (NPOS * 9)        // 21546
#define NSORT 32768
#define PRE  2000
#define POST 300

__constant__ float c_AB[9][4] = {
    {-37.254833f, -82.50967f,   53.254833f,  98.50967f},
    {-82.50967f,  -173.01933f,  98.50967f,  189.01933f},
    {-173.01933f, -354.03867f, 189.01933f,  370.03867f},
    {-56.f,       -56.f,        72.f,        72.f},
    {-120.f,      -120.f,      136.f,       136.f},
    {-248.f,      -248.f,      264.f,       264.f},
    {-82.50967f,  -37.254833f,  98.50967f,   53.254833f},
    {-173.01933f, -82.50967f,  189.01933f,   98.50967f},
    {-354.03867f, -173.01933f, 370.03867f,  189.01933f}};

// ---------------------------------------------------------------------------
// WMMA bf16 GEMM:  C[M,N] = act( A[M,K] * B + bias )
//   bT==0 : B is [K,N] row-major        bT==1 : B is [N,K] row-major
// Tile: 64(M) x 128(N) per 256-thread block (8 waves, each 32x32 = 2x2 WMMA).
// Double-buffered LDS (one barrier per K-tile); f32 -> bf16 conversion is
// packed into dwords so staging emits single ds_store_b128 per 8 values.
// (All GEMMs here are HBM-bound: FC1 AI = 150 flop/byte << roofline, so the
// only job of the staging path is to stay off the critical bandwidth path.)
// ---------------------------------------------------------------------------
__global__ void __launch_bounds__(256) gemm_bf16_wmma(
    const float* __restrict__ A, const float* __restrict__ B,
    const float* __restrict__ bias, float* __restrict__ C,
    int M, int N, int K, int bT, int relu)
{
    __shared__ __align__(16) unsigned short sA[2][64][40];   // [m][k], stride 80B
    __shared__ __align__(16) unsigned short sB[2][128][40];  // [n][k], stride 80B

    const int tid  = threadIdx.x;
    const int m0   = blockIdx.y * 64;
    const int n0   = blockIdx.x * 128;
    const int wid  = tid >> 5, lane = tid & 31;
    const int wm   = wid >> 2, wn = wid & 3;      // 2x4 wave grid
    const int half = lane >> 4, lm = lane & 15;

    v8f acc[2][2] = {};

    const int arow  = tid >> 2;            // 64 rows, 4 threads/row
    const int akoff = (tid & 3) << 3;      // 8 floats each
    const int nK = K >> 5;                 // all K here are multiples of 32

    // ---- stage one 64x32 A tile + 32x128 B tile into LDS buffer `buf` ----
    auto stage = [&](int buf, int k0) {
        {   // A tile, packed dword stores -> ds_store_b128
            const int gm = m0 + arow;
            uint4 pk = make_uint4(0u, 0u, 0u, 0u);
            if (gm < M) {
                const float4* p = (const float4*)(A + (size_t)gm * K + k0 + akoff);
                float4 v0 = p[0], v1 = p[1];
                pk.x = pack2bf(v0.x, v0.y); pk.y = pack2bf(v0.z, v0.w);
                pk.z = pack2bf(v1.x, v1.y); pk.w = pack2bf(v1.z, v1.w);
                // prefetch tile after next (global_prefetch_b8)
                if (k0 + 64 < K)
                    __builtin_prefetch(A + (size_t)gm * K + k0 + 32 + akoff, 0, 1);
            }
            *(uint4*)&sA[buf][arow][akoff] = pk;
        }
        if (bT) {   // B[N,K]: contiguous K runs, packed dword stores
            const int bn = tid >> 1, bk = (tid & 1) << 4;  // 16 along K
            const int gn = n0 + bn;
            uint4 p0 = make_uint4(0u, 0u, 0u, 0u);
            uint4 p1 = make_uint4(0u, 0u, 0u, 0u);
            if (gn < N) {
                const float4* p = (const float4*)(B + (size_t)gn * K + k0 + bk);
                float4 v0 = p[0], v1 = p[1], v2 = p[2], v3 = p[3];
                p0.x = pack2bf(v0.x, v0.y); p0.y = pack2bf(v0.z, v0.w);
                p0.z = pack2bf(v1.x, v1.y); p0.w = pack2bf(v1.z, v1.w);
                p1.x = pack2bf(v2.x, v2.y); p1.y = pack2bf(v2.z, v2.w);
                p1.z = pack2bf(v3.x, v3.y); p1.w = pack2bf(v3.z, v3.w);
                if (k0 + 64 < K)
                    __builtin_prefetch(B + (size_t)gn * K + k0 + 32 + bk, 0, 1);
            }
            *(uint4*)&sB[buf][bn][bk]     = p0;
            *(uint4*)&sB[buf][bn][bk + 8] = p1;
        } else {    // B[K,N]: coalesced global reads, transposed LDS stores
            const int bk = tid >> 3, bnoff = (tid & 7) << 4;  // 16 along N
            const float* p = B + (size_t)(k0 + bk) * N + n0 + bnoff;
            #pragma unroll
            for (int e = 0; e < 16; ++e) {
                const int gn = n0 + bnoff + e;
                sB[buf][bnoff + e][bk] = (gn < N) ? f2bf(p[e]) : (unsigned short)0;
            }
            if (k0 + 64 < K)
                __builtin_prefetch(B + (size_t)(k0 + 32 + bk) * N + n0 + bnoff, 0, 1);
        }
    };

    stage(0, 0);

    for (int kt = 0; kt < nK; ++kt) {
        __syncthreads();                       // staged buffer visible / prev reads done
        const int cur = kt & 1;
        if (kt + 1 < nK) stage(cur ^ 1, (kt + 1) << 5);

        #pragma unroll
        for (int i = 0; i < 2; ++i) {
            // A fragment (ISA 7.12.2: lane=row, K 0-7|8-15 then 16-23|24-31 by half)
            const int ar = wm * 32 + i * 16 + lm;
            v8bf alo = *(const v8bf*)&sA[cur][ar][half * 8];
            v8bf ahi = *(const v8bf*)&sA[cur][ar][16 + half * 8];
            v16bf af;
            #pragma unroll
            for (int e = 0; e < 8; ++e) { af[e] = alo[e]; af[8 + e] = ahi[e]; }

            #pragma unroll
            for (int j = 0; j < 2; ++j) {
                // B fragment: lane=col, K 0-7|8-15 then 16-23|24-31 by half (same as A)
                const int bn = wn * 32 + j * 16 + lm;
                v8bf blo = *(const v8bf*)&sB[cur][bn][half * 8];
                v8bf bhi = *(const v8bf*)&sB[cur][bn][16 + half * 8];
                v16bf bfm;
                #pragma unroll
                for (int e = 0; e < 8; ++e) { bfm[e] = blo[e]; bfm[8 + e] = bhi[e]; }

                acc[i][j] = __builtin_amdgcn_wmma_f32_16x16x32_bf16(
                    false, af, false, bfm, (short)0, acc[i][j], false, false);
            }
        }
    }

    // ---- store (D layout: m = vgpr + 8*half, n = lane&15), ldc == N ----
    #pragma unroll
    for (int i = 0; i < 2; ++i)
        #pragma unroll
        for (int j = 0; j < 2; ++j)
            #pragma unroll
            for (int r = 0; r < 8; ++r) {
                const int m = m0 + wm * 32 + i * 16 + half * 8 + r;
                const int n = n0 + wn * 32 + j * 16 + lm;
                if (m < M && n < N) {
                    float v = acc[i][j][r];
                    if (bias) v += bias[n];
                    if (relu) v = fmaxf(v, 0.f);
                    C[(size_t)m * N + n] = v;
                }
            }
}

// ---------------------------------------------------------------------------
// im2col for the stride-16 16x16 VALID extractor conv: [2394, 768]
// col = c*256 + ky*16 + kx  (matches Wext [512][3][16][16])
// ---------------------------------------------------------------------------
__global__ void im2col_ext(const float* __restrict__ x, float* __restrict__ out)
{
    const int t = blockIdx.x * blockDim.x + threadIdx.x;
    if (t >= NPOS * 768) return;
    const int row = t / 768, col = t % 768;
    const int c = col >> 8, rem = col & 255, ky = rem >> 4, kx = rem & 15;
    const int oy = row / FW, ox = row % FW;
    out[t] = x[(size_t)c * HIMG * WIMG + (oy * 16 + ky) * WIMG + (ox * 16 + kx)];
}

// ---------------------------------------------------------------------------
// im2col for 3x3 SAME conv over featHWC [2394,512]: [2394, 4608]
// col = c*9 + ky*3 + kx  (matches Wc1 [512][512][3][3])
// ---------------------------------------------------------------------------
__global__ void im2col_3x3(const float* __restrict__ feat, float* __restrict__ out)
{
    const int t = blockIdx.x * blockDim.x + threadIdx.x;
    if (t >= NPOS * 4608) return;
    const int row = t / 4608, col = t % 4608;
    const int c = col / 9, rem = col % 9, ky = rem / 3, kx = rem % 3;
    const int y = row / FW, x = row % FW;
    const int iy = y + ky - 1, ix = x + kx - 1;
    float v = 0.f;
    if (iy >= 0 && iy < FH && ix >= 0 && ix < FW)
        v = feat[(size_t)(iy * FW + ix) * 512 + c];
    out[t] = v;
}

// ---------------------------------------------------------------------------
// Proposal decode: anchors + loc2bbox + clip + min-size + score/key setup
// ---------------------------------------------------------------------------
__global__ void proposals_kernel(const float* __restrict__ rpnloc,
                                 const float* __restrict__ rpnsc,
                                 const int*   __restrict__ scale_p,
                                 float* __restrict__ boxes, int* __restrict__ validf,
                                 float* __restrict__ keys, int* __restrict__ vals)
{
    const int a = blockIdx.x * blockDim.x + threadIdx.x;
    if (a >= NSORT) return;
    if (a >= NANCH) { keys[a] = -__builtin_inff(); vals[a] = -1; return; }

    const int cell = a / 9, i = a % 9;
    const int gy = cell / FW, gx = cell % FW;
    const float sy = gy * 16.f, sx = gx * 16.f;
    const float a0 = c_AB[i][0] + sy, a1 = c_AB[i][1] + sx;
    const float a2 = c_AB[i][2] + sy, a3 = c_AB[i][3] + sx;

    const float* L = rpnloc + (size_t)cell * 36 + i * 4;
    const float ph = a2 - a0, pw = a3 - a1;
    const float py = a0 + 0.5f * ph, px = a1 + 0.5f * pw;
    const float cy = L[0] * ph + py, cx = L[1] * pw + px;
    const float hh = expf(L[2]) * ph, ww = expf(L[3]) * pw;

    float y0 = cy - 0.5f * hh, x0 = cx - 0.5f * ww;
    float y1 = cy + 0.5f * hh, x1 = cx + 0.5f * ww;
    y0 = fminf(fmaxf(y0, 0.f), (float)HIMG); x0 = fminf(fmaxf(x0, 0.f), (float)WIMG);
    y1 = fminf(fmaxf(y1, 0.f), (float)HIMG); x1 = fminf(fmaxf(x1, 0.f), (float)WIMG);

    const float ms = 16.f * (float)scale_p[0];
    const int valid = ((y1 - y0) >= ms) && ((x1 - x0) >= ms);

    boxes[a * 4 + 0] = y0; boxes[a * 4 + 1] = x0;
    boxes[a * 4 + 2] = y1; boxes[a * 4 + 3] = x1;
    validf[a] = valid;
    keys[a]   = valid ? rpnsc[(size_t)cell * 18 + i * 2 + 1] : -__builtin_inff();
    vals[a]   = a;
}

// one (k,j) stage of a descending bitonic sort over 32768 key/idx pairs
__global__ void bitonic_step(float* __restrict__ keys, int* __restrict__ vals,
                             int j, int k)
{
    const int i = blockIdx.x * blockDim.x + threadIdx.x;
    const int p = i ^ j;
    if (p > i) {
        const float ki = keys[i], kp = keys[p];
        const bool up = ((i & k) == 0);
        const bool sw = up ? (ki < kp) : (ki > kp);   // descending overall
        if (sw) {
            keys[i] = kp; keys[p] = ki;
            const int vi = vals[i]; vals[i] = vals[p]; vals[p] = vi;
        }
    }
}

__global__ void gather_top(const float* __restrict__ boxes,
                           const int* __restrict__ validf,
                           const int* __restrict__ vals,
                           float* __restrict__ boxes_s, int* __restrict__ vb)
{
    const int t = blockIdx.x * blockDim.x + threadIdx.x;
    if (t >= PRE) return;
    const int id = vals[t];
    if (id >= 0) {
        boxes_s[t * 4 + 0] = boxes[id * 4 + 0];
        boxes_s[t * 4 + 1] = boxes[id * 4 + 1];
        boxes_s[t * 4 + 2] = boxes[id * 4 + 2];
        boxes_s[t * 4 + 3] = boxes[id * 4 + 3];
        vb[t] = validf[id];
    } else {
        boxes_s[t * 4 + 0] = 0.f; boxes_s[t * 4 + 1] = 0.f;
        boxes_s[t * 4 + 2] = 0.f; boxes_s[t * 4 + 3] = 0.f;
        vb[t] = 0;
    }
}

// ---------------------------------------------------------------------------
// Greedy sequential NMS (boxes sorted desc) + stable keep-first compaction.
// Single workgroup; boxes/areas/flags live in LDS (48KB).
// ---------------------------------------------------------------------------
__global__ void __launch_bounds__(256) nms_compact(
    const float* __restrict__ boxes_s, const int* __restrict__ vb,
    float* __restrict__ rois_ws, float* __restrict__ rois_out)
{
    __shared__ float4 sb[PRE];
    __shared__ float  sa[PRE];
    __shared__ int    sk[PRE];
    const int t = threadIdx.x;

    for (int i = t; i < PRE; i += 256) {
        float4 b = ((const float4*)boxes_s)[i];
        sb[i] = b;
        sa[i] = (b.z - b.x + 1.f) * (b.w - b.y + 1.f);
        sk[i] = vb[i];
    }
    __syncthreads();

    for (int i = 0; i < PRE; ++i) {
        if (sk[i]) {                          // uniform (LDS) -> no divergence hazard
            const float4 b = sb[i];
            const float  ai = sa[i];
            for (int j = i + 1 + t; j < PRE; j += 256) {
                if (sk[j]) {
                    const float iy1 = fmaxf(b.x, sb[j].x), ix1 = fmaxf(b.y, sb[j].y);
                    const float iy2 = fminf(b.z, sb[j].z), ix2 = fminf(b.w, sb[j].w);
                    const float inter = fmaxf(iy2 - iy1 + 1.f, 0.f) *
                                        fmaxf(ix2 - ix1 + 1.f, 0.f);
                    const float iou = inter / (ai + sa[j] - inter);
                    if (iou > 0.7f) sk[j] = 0;
                }
            }
        }
        __syncthreads();
    }

    if (t == 0) {  // kept boxes first (index order), then un-kept, take 300
        int cnt = 0;
        for (int pass = 0; pass < 2 && cnt < POST; ++pass)
            for (int i = 0; i < PRE && cnt < POST; ++i)
                if ((pass == 0) ? (sk[i] != 0) : (sk[i] == 0)) {
                    const float4 b = sb[i];
                    rois_ws[cnt * 4 + 0] = b.x; rois_ws[cnt * 4 + 1] = b.y;
                    rois_ws[cnt * 4 + 2] = b.z; rois_ws[cnt * 4 + 3] = b.w;
                    rois_out[cnt * 4 + 0] = b.x; rois_out[cnt * 4 + 1] = b.y;
                    rois_out[cnt * 4 + 2] = b.z; rois_out[cnt * 4 + 3] = b.w;
                    ++cnt;
                }
    }
}

// ---------------------------------------------------------------------------
// Adaptive 7x7 ROI max-pool over featHWC [(y*63+x)*512 + c] -> pool[300,25088]
// out col = c*49 + by*7 + bx  (matches [C,7,7] flatten)
// ---------------------------------------------------------------------------
__global__ void roi_pool(const float* __restrict__ feat,
                         const float* __restrict__ rois,
                         float* __restrict__ pool)
{
    const int r = blockIdx.x;
    const float* b = rois + r * 4;
    const int y0 = min(max((int)floorf(b[0] * (1.f / 16.f)), 0), FH - 1);
    const int x0 = min(max((int)floorf(b[1] * (1.f / 16.f)), 0), FW - 1);
    const int y1 = min(max((int)floorf(b[2] * (1.f / 16.f)), 0), FH - 1);
    const int x1 = min(max((int)floorf(b[3] * (1.f / 16.f)), 0), FW - 1);
    const int H = y1 - y0 + 1, W = x1 - x0 + 1;

    for (int o = threadIdx.x; o < 25088; o += blockDim.x) {
        const int c = o / 49, rem = o % 49, by = rem / 7, bx = rem % 7;
        const int rs = y0 + (by * H) / 7,  re = y0 + ((by + 1) * H + 6) / 7;
        const int cs = x0 + (bx * W) / 7,  ce = x0 + ((bx + 1) * W + 6) / 7;
        float m = -1e30f;
        for (int y = rs; y < re; ++y)
            for (int x = cs; x < ce; ++x)
                m = fmaxf(m, feat[(size_t)(y * FW + x) * 512 + c]);
        pool[(size_t)r * 25088 + o] = m;
    }
}

// ---------------------------------------------------------------------------
// Host-side orchestration
// ---------------------------------------------------------------------------
extern "C" void kernel_launch(void* const* d_in, const int* in_sizes, int n_in,
                              void* d_out, int out_size, void* d_ws, size_t ws_size,
                              hipStream_t stream)
{
    (void)in_sizes; (void)n_in; (void)out_size; (void)ws_size;
    const float* x    = (const float*)d_in[0];
    const float* Wext = (const float*)d_in[1];
    const float* Wc1  = (const float*)d_in[2];
    const float* bc1  = (const float*)d_in[3];
    const float* Wloc = (const float*)d_in[4];
    const float* bloc = (const float*)d_in[5];
    const float* Wsc  = (const float*)d_in[6];
    const float* bsc  = (const float*)d_in[7];
    const float* W1   = (const float*)d_in[8];
    const float* b1   = (const float*)d_in[9];
    const float* W2   = (const float*)d_in[10];
    const float* b2   = (const float*)d_in[11];
    const float* Wl   = (const float*)d_in[12];
    const float* bl   = (const float*)d_in[13];
    const float* Ws   = (const float*)d_in[14];
    const float* bs   = (const float*)d_in[15];
    const int*   scale = (const int*)d_in[16];
    float* out = (float*)d_out;   // [300*84 | 300*21 | 300*4]

    char* base = (char*)d_ws;
    size_t off = 0;
    auto alloc = [&](size_t bytes) -> void* {
        void* p = base + off;
        off = (off + bytes + 255) & ~(size_t)255;
        return p;
    };
    // big regions (aliased across pipeline phases)
    float* bufA    = (float*)alloc((size_t)NPOS * 768 * 4);   // im2col_ext, later fc6 (300*4096)
    float* featHWC = (float*)alloc((size_t)NPOS * 512 * 4);
    float* bufB    = (float*)alloc((size_t)NPOS * 4608 * 4);  // im2col_3x3, later pool (300*25088)
    float* layer1  = (float*)alloc((size_t)NPOS * 512 * 4);
    float* fc7     = (float*)alloc((size_t)POST * 4096 * 4);
    float* rpnloc  = (float*)alloc((size_t)NPOS * 36 * 4);
    float* rpnsc   = (float*)alloc((size_t)NPOS * 18 * 4);
    float* boxes   = (float*)alloc((size_t)NANCH * 4 * 4);
    int*   validf  = (int*)  alloc((size_t)NANCH * 4);
    float* keys    = (float*)alloc((size_t)NSORT * 4);
    int*   vals    = (int*)  alloc((size_t)NSORT * 4);
    float* boxes_s = (float*)alloc((size_t)PRE * 4 * 4);
    int*   vb      = (int*)  alloc((size_t)PRE * 4);
    float* rois_ws = (float*)alloc((size_t)POST * 4 * 4);

    // 1) extractor conv as GEMM: feat[2394,512] = relu(im2colE[2394,768] * Wext[512,768]^T)
    im2col_ext<<<(NPOS * 768 + 255) / 256, 256, 0, stream>>>(x, bufA);
    gemm_bf16_wmma<<<dim3(4, (NPOS + 63) / 64), 256, 0, stream>>>(
        bufA, Wext, nullptr, featHWC, NPOS, 512, 768, 1, 1);

    // 2) 3x3 conv: layer1 = relu(im2col3[2394,4608] * Wc1[512,4608]^T + bc1)
    im2col_3x3<<<((size_t)NPOS * 4608 + 255) / 256, 256, 0, stream>>>(featHWC, bufB);
    gemm_bf16_wmma<<<dim3(4, (NPOS + 63) / 64), 256, 0, stream>>>(
        bufB, Wc1, bc1, layer1, NPOS, 512, 4608, 1, 1);

    // 3) RPN 1x1 convs
    gemm_bf16_wmma<<<dim3(1, (NPOS + 63) / 64), 256, 0, stream>>>(
        layer1, Wloc, bloc, rpnloc, NPOS, 36, 512, 1, 0);
    gemm_bf16_wmma<<<dim3(1, (NPOS + 63) / 64), 256, 0, stream>>>(
        layer1, Wsc, bsc, rpnsc, NPOS, 18, 512, 1, 0);

    // 4) decode proposals, sort by score (desc), top-2000
    proposals_kernel<<<NSORT / 256, 256, 0, stream>>>(
        rpnloc, rpnsc, scale, boxes, validf, keys, vals);
    for (int k = 2; k <= NSORT; k <<= 1)
        for (int j = k >> 1; j > 0; j >>= 1)
            bitonic_step<<<NSORT / 256, 256, 0, stream>>>(keys, vals, j, k);
    gather_top<<<(PRE + 255) / 256, 256, 0, stream>>>(boxes, validf, vals, boxes_s, vb);

    // 5) NMS + select 300 ROIs (also written to d_out tail)
    nms_compact<<<1, 256, 0, stream>>>(boxes_s, vb, rois_ws,
                                       out + POST * 84 + POST * 21);

    // 6) ROI max-pool -> pool[300,25088] (aliases bufB)
    roi_pool<<<POST, 256, 0, stream>>>(featHWC, rois_ws, bufB);

    // 7) FC head (HBM-bound on W1/W2 stream; bf16 WMMA, f32 accumulate)
    gemm_bf16_wmma<<<dim3(32, (POST + 63) / 64), 256, 0, stream>>>(
        bufB, W1, b1, bufA, POST, 4096, 25088, 0, 1);          // fc6 (aliases bufA)
    gemm_bf16_wmma<<<dim3(32, (POST + 63) / 64), 256, 0, stream>>>(
        bufA, W2, b2, fc7, POST, 4096, 4096, 0, 1);            // fc7
    gemm_bf16_wmma<<<dim3(1, (POST + 63) / 64), 256, 0, stream>>>(
        fc7, Wl, bl, out, POST, 84, 4096, 0, 0);               // class locs
    gemm_bf16_wmma<<<dim3(1, (POST + 63) / 64), 256, 0, stream>>>(
        fc7, Ws, bs, out + POST * 84, POST, 21, 4096, 0, 0);   // class scores
}
